// GRUModel_30623116821141
// MI455X (gfx1250) — matrix operand order repaired and checked by reading, not compiled, not run on hardware
//
#include <hip/hip_runtime.h>
#include <hip/hip_fp16.h>
#include <hip/hip_bf16.h>
#include <stdint.h>

typedef __attribute__((ext_vector_type(16))) _Float16 v16h;
typedef __attribute__((ext_vector_type(8)))  _Float16 v8h;
typedef __attribute__((ext_vector_type(8)))  float    v8f;

#define T_STEPS 512
#define BATCH   128
#define NIN     64
#define NHID    1024
#define NOUT    10

#define NWG     64          // == NT: one workgroup per 16-column output strip
#define TPB     256         // 8 wave32 per WG; wave w owns M-tile mt = w
#define WAVES_PER_WG (TPB/32)
#define MT      (BATCH/16)  // 8 M tiles
#define NT      (NHID/16)   // 64 N tiles
#define KT_H    (NHID/32)   // 32 K tiles (recurrent path)
#define KT_U    (NIN/32)    // 2  K tiles (input path)

// per-WG LDS slice sizes, in halves (fp16 elements)
#define SL_H    (KT_H*32*16)    // 16384 halves = 32 KB   (per h-weight)
#define SL_U    (KT_U*32*16)    // 1024 halves  = 2 KB    (per u-weight)
#define SMEM_HALVES (3*SL_H + 3*SL_U)          // 52224
#define SMEM_BYTES  (SMEM_HALVES*2)            // 104448 B < 320 KB/WGP

// ---------------------------------------------------------------- utilities

__global__ void zero_b32(uint32_t* p, size_t n_dwords) {
    size_t i = (size_t)blockIdx.x * blockDim.x + threadIdx.x;
    if (i < n_dwords) p[i] = 0u;
}

__global__ void cvt_f32_to_f16(const float* __restrict__ src,
                               _Float16* __restrict__ dst, size_t n) {
    size_t i = (size_t)blockIdx.x * blockDim.x + threadIdx.x;
    if (i < n) dst[i] = (_Float16)src[i];
}

// Pack a row-major (K x NHID) fp32 weight into per-lane WMMA B-fragment
// order in fp16:  pk[((nt*KT + kt)*32 + lane)*16 + j]  with
//   K index = kt*32 + (lane>>4)*16 + j,  N index = nt*16 + (lane&15)
// nt-major order => each WG's slice is one contiguous 32KB (2KB) block.
__global__ void pack_w16(const float* __restrict__ W,
                         _Float16* __restrict__ pk, int K) {
    int KTt   = K / 32;
    int tid   = blockIdx.x * blockDim.x + threadIdx.x;
    int total = KTt * NT * 32;
    if (tid >= total) return;
    int lane  = tid & 31;
    int tile  = tid >> 5;          // tile = nt*KTt + kt
    int kt    = tile % KTt;
    int nt    = tile / KTt;
    int n     = nt * 16 + (lane & 15);
    int kbase = kt * 32 + (lane >> 4) * 16;
    _Float16* dst = pk + (size_t)tid * 16;
#pragma unroll
    for (int j = 0; j < 16; ++j)
        dst[j] = (_Float16)W[(size_t)(kbase + j) * NHID + n];
}

// ---------------------------------------------------------------- persistent GRU

struct GruParams {
    const float* b_z; const float* b_r; const float* b_h;
    const float* W_hy; const float* b_y;
    float*    h32;   // master hidden state fp32 (B x NHID)
    _Float16* h16;   // fp16 copy of h for WMMA A loads
    _Float16* rh16;  // r * h, fp16, for phase 2 A loads
    float*    z32;   // update gate, fp32
    const _Float16* u16;                       // (T x B x NIN) fp16
    const _Float16* pk_hz; const _Float16* pk_hr; const _Float16* pk_hh;
    const _Float16* pk_uz; const _Float16* pk_ur; const _Float16* pk_uh;
    unsigned* bar;   // [0]=arrive counter, [1]=generation
    float*    out;   // d_out: logits (B*NOUT) then h_final (B*NHID)
};

__device__ __forceinline__ void grid_barrier(unsigned* bar) {
    __syncthreads();
    if (threadIdx.x == 0) {
        __threadfence();  // release prior global writes device-wide
        unsigned g = __hip_atomic_load(&bar[1], __ATOMIC_RELAXED,
                                       __HIP_MEMORY_SCOPE_AGENT);
        unsigned a = __hip_atomic_fetch_add(&bar[0], 1u, __ATOMIC_ACQ_REL,
                                            __HIP_MEMORY_SCOPE_AGENT);
        if (a == NWG - 1) {
            __hip_atomic_store(&bar[0], 0u, __ATOMIC_RELAXED,
                               __HIP_MEMORY_SCOPE_AGENT);
            __hip_atomic_store(&bar[1], g + 1u, __ATOMIC_RELEASE,
                               __HIP_MEMORY_SCOPE_AGENT);
        } else {
            while (__hip_atomic_load(&bar[1], __ATOMIC_ACQUIRE,
                                     __HIP_MEMORY_SCOPE_AGENT) == g)
                __builtin_amdgcn_s_sleep(1);
        }
    }
    __syncthreads();
}

// A fragment from a row-major fp16 matrix (two contiguous 8-half runs)
__device__ __forceinline__ v16h loadA(const _Float16* __restrict__ arow,
                                      int kt, int hsel) {
    union { v16h v; v8h h[2]; } A;
    A.h[0] = *(const v8h*)(arow + kt * 32 + hsel * 16);
    A.h[1] = *(const v8h*)(arow + kt * 32 + hsel * 16 + 8);
    return A.v;
}

// B fragment from the WG's LDS-resident packed weight slice
__device__ __forceinline__ v16h loadB_lds(const _Float16* __restrict__ s,
                                          int kt, int lane) {
    union { v16h v; v8h h[2]; } B;
    const v8h* bp = (const v8h*)(s + ((kt * 32 + lane) << 4));
    B.h[0] = bp[0];
    B.h[1] = bp[1];
    return B.v;
}

__device__ __forceinline__ v8f wmma(v16h a, v16h b, v8f c) {
    return __builtin_amdgcn_wmma_f32_16x16x32_f16(
        false, a, false, b, (short)0, c, false, false);
}

// stage one contiguous packed-weight slice into LDS (all 256 threads)
__device__ __forceinline__ void stage(const _Float16* __restrict__ g,
                                      _Float16* __restrict__ s,
                                      int len_halves) {
    const uint4* gs = (const uint4*)g;
    uint4*       ss = (uint4*)s;
    int n = len_halves / 8;
    for (int i = threadIdx.x; i < n; i += TPB) ss[i] = gs[i];
}

__global__ __launch_bounds__(TPB, 1) void gru_persistent(GruParams p) {
    extern __shared__ _Float16 smem[];
    _Float16* s_hz = smem;
    _Float16* s_hr = smem + SL_H;
    _Float16* s_hh = smem + 2 * SL_H;
    _Float16* s_uz = smem + 3 * SL_H;
    _Float16* s_ur = smem + 3 * SL_H + SL_U;
    _Float16* s_uh = smem + 3 * SL_H + 2 * SL_U;

    const int lane = threadIdx.x & 31;
    const int wave = threadIdx.x >> 5;     // == mt
    const int nt   = blockIdx.x;           // NWG == NT
    const int mt   = wave;
    const int hsel = lane >> 4;
    const int col0 = lane & 15;
    const int m    = mt * 16 + col0;       // A-row this lane reads
    const int col  = nt * 16 + col0;       // D-column this lane owns

    // ---- one-time: pin this WG's weight slices in LDS (reused 512 steps)
    stage(p.pk_hz + (size_t)nt * SL_H, s_hz, SL_H);
    stage(p.pk_hr + (size_t)nt * SL_H, s_hr, SL_H);
    stage(p.pk_hh + (size_t)nt * SL_H, s_hh, SL_H);
    stage(p.pk_uz + (size_t)nt * SL_U, s_uz, SL_U);
    stage(p.pk_ur + (size_t)nt * SL_U, s_ur, SL_U);
    stage(p.pk_uh + (size_t)nt * SL_U, s_uh, SL_U);
    __syncthreads();

    const _Float16* hrow  = p.h16  + (size_t)m * NHID;
    const _Float16* rhrow = p.rh16 + (size_t)m * NHID;
    const float bz = p.b_z[col];
    const float br = p.b_r[col];
    const float bh = p.b_h[col];

    for (int t = 0; t < T_STEPS; ++t) {
        const _Float16* urow = p.u16 + ((size_t)t * BATCH + m) * NIN;

        // ---- phase 1: z and r gates together (A fragment shared, 2 WMMA/kt)
        {
            v8f az = {}, ar = {};
            for (int kt = 0; kt < KT_H; ++kt) {
                v16h A = loadA(hrow, kt, hsel);
                az = wmma(A, loadB_lds(s_hz, kt, lane), az);
                ar = wmma(A, loadB_lds(s_hr, kt, lane), ar);
            }
            for (int kt = 0; kt < KT_U; ++kt) {
                v16h A = loadA(urow, kt, hsel);
                az = wmma(A, loadB_lds(s_uz, kt, lane), az);
                ar = wmma(A, loadB_lds(s_ur, kt, lane), ar);
            }
#pragma unroll
            for (int v = 0; v < 8; ++v) {
                int    row = mt * 16 + v + 8 * hsel;
                size_t idx = (size_t)row * NHID + col;
                float  z   = 1.f / (1.f + __expf(-(az[v] + bz)));
                float  r   = 1.f / (1.f + __expf(-(ar[v] + br)));
                p.z32[idx]  = z;
                p.rh16[idx] = (_Float16)(r * p.h32[idx]);
            }
        }
        grid_barrier(p.bar);

        // ---- phase 2: candidate h~ and state update
        {
            v8f ah = {};
            for (int kt = 0; kt < KT_H; ++kt)
                ah = wmma(loadA(rhrow, kt, hsel), loadB_lds(s_hh, kt, lane), ah);
            for (int kt = 0; kt < KT_U; ++kt)
                ah = wmma(loadA(urow, kt, hsel), loadB_lds(s_uh, kt, lane), ah);
#pragma unroll
            for (int v = 0; v < 8; ++v) {
                int    row = mt * 16 + v + 8 * hsel;
                size_t idx = (size_t)row * NHID + col;
                float  ht  = tanhf(ah[v] + bh);
                float  z   = p.z32[idx];
                float  hn  = (1.f - z) * p.h32[idx] + z * ht;
                p.h32[idx] = hn;
                p.h16[idx] = (_Float16)hn;
            }
        }
        grid_barrier(p.bar);
    }

    // ---- epilogue: logits = h @ W_hy + b_y (tiny, fp32 VALU), copy h_final
    int tid = blockIdx.x * TPB + threadIdx.x;
    if (tid < BATCH * NOUT) {
        int   b = tid / NOUT, o = tid % NOUT;
        float s = p.b_y[o];
        for (int k = 0; k < NHID; ++k)
            s += p.h32[(size_t)b * NHID + k] * p.W_hy[(size_t)k * NOUT + o];
        p.out[tid] = s;
    }
    for (size_t i = (size_t)tid; i < (size_t)BATCH * NHID; i += (size_t)NWG * TPB)
        p.out[BATCH * NOUT + i] = p.h32[i];
}

// ---------------------------------------------------------------- launch

extern "C" void kernel_launch(void* const* d_in, const int* in_sizes, int n_in,
                              void* d_out, int out_size, void* d_ws, size_t ws_size,
                              hipStream_t stream) {
    (void)in_sizes; (void)n_in; (void)out_size; (void)ws_size;

    const float* u    = (const float*)d_in[0];
    const float* W_uz = (const float*)d_in[1];
    const float* W_hz = (const float*)d_in[2];
    const float* b_z  = (const float*)d_in[3];
    const float* W_ur = (const float*)d_in[4];
    const float* W_hr = (const float*)d_in[5];
    const float* b_r  = (const float*)d_in[6];
    const float* W_uh = (const float*)d_in[7];
    const float* W_hh = (const float*)d_in[8];
    const float* b_h  = (const float*)d_in[9];
    const float* W_hy = (const float*)d_in[10];
    const float* b_y  = (const float*)d_in[11];

    uint8_t* ws = (uint8_t*)d_ws;
    size_t o = 0;
    auto take = [&](size_t bytes) { size_t r = o; o += (bytes + 255) & ~(size_t)255; return r; };

    size_t off_bar  = take(256);
    size_t off_h32  = take((size_t)BATCH * NHID * 4);
    size_t off_h16  = take((size_t)BATCH * NHID * 2);
    size_t zero_end = o;                         // zero [0, zero_end): bar,h32,h16
    size_t off_rh16 = take((size_t)BATCH * NHID * 2);
    size_t off_z32  = take((size_t)BATCH * NHID * 4);
    size_t off_u16  = take((size_t)T_STEPS * BATCH * NIN * 2);
    size_t off_phz  = take((size_t)NHID * NHID * 2);
    size_t off_phr  = take((size_t)NHID * NHID * 2);
    size_t off_phh  = take((size_t)NHID * NHID * 2);
    size_t off_puz  = take((size_t)NIN * NHID * 2);
    size_t off_pur  = take((size_t)NIN * NHID * 2);
    size_t off_puh  = take((size_t)NIN * NHID * 2);

    // init: zero barrier + h state (fresh every launch -> deterministic)
    {
        size_t nd = zero_end / 4;
        zero_b32<<<(unsigned)((nd + 255) / 256), 256, 0, stream>>>((uint32_t*)ws, nd);
    }
    // convert u to fp16 (row-major is already WMMA-A friendly)
    {
        size_t n = (size_t)T_STEPS * BATCH * NIN;
        cvt_f32_to_f16<<<(unsigned)((n + 255) / 256), 256, 0, stream>>>(
            u, (_Float16*)(ws + off_u16), n);
    }
    // pack weights into B-fragment order (fp16), nt-major for LDS staging
    {
        int nh = (NHID / 32) * NT * 32;   // 65536 threads
        int nu = (NIN  / 32) * NT * 32;   // 4096 threads
        pack_w16<<<(nh + 255) / 256, 256, 0, stream>>>(W_hz, (_Float16*)(ws + off_phz), NHID);
        pack_w16<<<(nh + 255) / 256, 256, 0, stream>>>(W_hr, (_Float16*)(ws + off_phr), NHID);
        pack_w16<<<(nh + 255) / 256, 256, 0, stream>>>(W_hh, (_Float16*)(ws + off_phh), NHID);
        pack_w16<<<(nu + 255) / 256, 256, 0, stream>>>(W_uz, (_Float16*)(ws + off_puz), NIN);
        pack_w16<<<(nu + 255) / 256, 256, 0, stream>>>(W_ur, (_Float16*)(ws + off_pur), NIN);
        pack_w16<<<(nu + 255) / 256, 256, 0, stream>>>(W_uh, (_Float16*)(ws + off_puh), NIN);
    }

    GruParams p;
    p.b_z = b_z; p.b_r = b_r; p.b_h = b_h; p.W_hy = W_hy; p.b_y = b_y;
    p.h32  = (float*)(ws + off_h32);
    p.h16  = (_Float16*)(ws + off_h16);
    p.rh16 = (_Float16*)(ws + off_rh16);
    p.z32  = (float*)(ws + off_z32);
    p.u16  = (const _Float16*)(ws + off_u16);
    p.pk_hz = (const _Float16*)(ws + off_phz);
    p.pk_hr = (const _Float16*)(ws + off_phr);
    p.pk_hh = (const _Float16*)(ws + off_phh);
    p.pk_uz = (const _Float16*)(ws + off_puz);
    p.pk_ur = (const _Float16*)(ws + off_pur);
    p.pk_uh = (const _Float16*)(ws + off_puh);
    p.bar = (unsigned*)(ws + off_bar);
    p.out = (float*)d_out;

    gru_persistent<<<NWG, TPB, SMEM_BYTES, stream>>>(p);
}
